// LSTM_8392366096809
// MI455X (gfx1250) — matrix operand order repaired and checked by reading, not compiled, not run on hardware
//
#include <hip/hip_runtime.h>
#include <hip/hip_bf16.h>

// ---------------------------------------------------------------------------
// Fused LSTM scan for MI455X (gfx1250), round 4.
//   - bf16 WMMA (v_wmma_f32_16x16x32_bf16), f32 accumulators, dual M tiles
//   - B (weight) fragments held in REGISTERS across the whole time loop:
//     loaded once from LDS, reused 512x -> eliminates 128 KB/step LDS traffic,
//     inner loop is pure WMMA + C stores
//   - gate bias in registers, folded into accumulator init
//   - hardware V_TANH_F32 for all activations
//   - 2 barriers/step: [GEMM] bar [elementwise + gather(t+1)] bar
// ---------------------------------------------------------------------------

typedef __attribute__((ext_vector_type(16))) __bf16 v16bf;
typedef __attribute__((ext_vector_type(8)))  __bf16 v8bf;
typedef __attribute__((ext_vector_type(8)))  float  v8f;

#define DIM_E   32
#define DIM_H   100
#define NGATE4  400      // 4 * DIM_H
#define KDIM    132      // DIM_E + DIM_H
#define KP      160      // K padded to multiple of 32
#define KB      5        // KP / 32
#define NT      25       // NGATE4 / 16
#define MT      2        // M tiles of 16 per workgroup
#define TILE_B  32       // MT * 16
#define THREADS 256      // 8 waves
#define NWAVE   8
#define TPW     4        // max register-resident N-tiles per wave (wave0: 4, rest: 3)
#define EPT     13       // ceil(TILE_B*DIM_H / THREADS)

// ---- hardware tanh (gfx1250 V_TANH_F32) ----
#if __has_builtin(__builtin_amdgcn_tanhf)
__device__ __forceinline__ float tanh_hw(float x) { return __builtin_amdgcn_tanhf(x); }
#else
__device__ __forceinline__ float tanh_hw(float x) {
    float y;
    asm volatile("v_tanh_f32 %0, %1\n\tv_nop" : "=v"(y) : "v"(x));
    return y;
}
#endif
__device__ __forceinline__ float sigmoid_hw(float x) {
    return __builtin_fmaf(0.5f, tanh_hw(0.5f * x), 0.5f);
}

__global__ __launch_bounds__(THREADS)
void lstm_fused_wmma(const int*   __restrict__ data,   // (B, T) token ids
                     const float* __restrict__ emb,    // (VOCAB, 32)
                     const float* __restrict__ Wf, const float* __restrict__ bf,
                     const float* __restrict__ Wi, const float* __restrict__ bi,
                     const float* __restrict__ Wo, const float* __restrict__ bo,
                     const float* __restrict__ Wc, const float* __restrict__ bc,
                     const float* __restrict__ Wy,     // (100, 1)
                     const float* __restrict__ by,     // (1,)
                     float* __restrict__ out,          // (B, 1)
                     int B, int T)
{
    extern __shared__ char smem[];
    __bf16* Wpack  = (__bf16*)smem;                       // NT*KB*512 bf16 = 128000 B
    __bf16* Asta   = Wpack + NT * KB * 512;               // TILE_B*KP bf16 =  10240 B
    float*  pre    = (float*)(Asta + TILE_B * KP);        // NT*MT*256 f32  =  51200 B
    float*  biasS  = pre + NT * MT * 256;                 // 400 f32
    float*  outAcc = biasS + NGATE4;                      // TILE_B f32
    int*    maskS  = (int*)(outAcc + TILE_B);             // 2*TILE_B
    int*    lastS  = maskS + 2 * TILE_B;                  // TILE_B

    const int tid  = threadIdx.x;
    const int wave = tid >> 5;
    const int lane = tid & 31;
    const int b0   = blockIdx.x * TILE_B;

    // ---- one-time: pack combined gate weights into B-fragment order ----
    // B operand (K x N = 32 x 16): lane<16 holds N=lane, K=kb*32+slot;
    // lane>=16 holds N=lane-16, K=kb*32+16+slot. 16 bf16 per lane contiguous.
    {
        const float* Wg[4] = {Wf, Wi, Wo, Wc};
        for (int idx = tid; idx < NT * KB * 512; idx += THREADS) {
            int slot = idx & 15;
            int l    = (idx >> 4) & 31;
            int g    = idx >> 9;                 // nt*KB + kb
            int kb   = g % KB;
            int nt   = g / KB;
            int k    = kb * 32 + ((l >> 4) << 4) + slot;
            int n    = nt * 16 + (l & 15);
            float v  = 0.0f;
            if (k < KDIM) v = Wg[n / DIM_H][k * DIM_H + (n % DIM_H)];
            Wpack[idx] = (__bf16)v;
        }
    }
    for (int n = tid; n < NGATE4; n += THREADS) {
        float v = (n < 100) ? bf[n] : (n < 200) ? bi[n - 100]
                : (n < 300) ? bo[n - 200] : bc[n - 300];
        biasS[n] = v;
    }
    for (int i = tid; i < TILE_B * KP; i += THREADS) Asta[i] = (__bf16)0.0f;
    if (tid < TILE_B) {
        int b = b0 + tid; if (b >= B) b = 0;
        lastS[tid]  = data[(long)b * T + (T - 1)];
        outAcc[tid] = 0.0f;
    }
    __syncthreads();   // Wpack/biasS/lastS ready

    // ---- hoist this wave's B fragments + biases into registers (reused T times)
    v16bf breg[TPW][KB];
    float bias_r[TPW];
    #pragma unroll
    for (int i = 0; i < TPW; ++i) {
        int nt = wave + i * NWAVE;
        if (nt < NT) {
            bias_r[i] = biasS[nt * 16 + (lane & 15)];
            #pragma unroll
            for (int kb = 0; kb < KB; ++kb) {
                const v8bf* q = (const v8bf*)&Wpack[(nt * KB + kb) * 512 + lane * 16];
                v8bf lo = q[0], hi = q[1];
                v16bf bb;
                #pragma unroll
                for (int s = 0; s < 8; ++s) { bb[s] = lo[s]; bb[s + 8] = hi[s]; }
                breg[i][kb] = bb;
            }
        } else {
            bias_r[i] = 0.0f;
            #pragma unroll
            for (int kb = 0; kb < KB; ++kb) breg[i][kb] = (v16bf)(__bf16)0.0f;
        }
    }

    // gather x(0) + mask(0) into buffer 0
    if (tid < TILE_B) {
        int b = b0 + tid; if (b >= B) b = 0;
        maskS[tid] = (data[(long)b * T] != lastS[tid]);
    }
    for (int idx = tid; idx < TILE_B * DIM_E; idx += THREADS) {
        int r = idx >> 5, c = idx & 31;
        int b = b0 + r; if (b >= B) b = 0;
        int tok = data[(long)b * T];
        Asta[r * KP + c] = (__bf16)emb[tok * DIM_E + c];
    }
    float c_reg[EPT], h_reg[EPT];
    #pragma unroll
    for (int i = 0; i < EPT; ++i) { c_reg[i] = 0.0f; h_reg[i] = 0.0f; }
    __syncthreads();

    // ------------------------------ time loop ------------------------------
    for (int t = 0; t < T; ++t) {
        // ---- GEMM: (32 x 160) @ (160 x 400); B already in registers ----
        v16bf afr[MT][KB];
        {
            int m     = lane & 15;
            int baseK = (lane >> 4) << 4;
            #pragma unroll
            for (int mt = 0; mt < MT; ++mt) {
                #pragma unroll
                for (int kb = 0; kb < KB; ++kb) {
                    const v8bf* p = (const v8bf*)&Asta[(mt * 16 + m) * KP + kb * 32 + baseK];
                    v8bf lo = p[0], hi = p[1];
                    v16bf a;
                    #pragma unroll
                    for (int s = 0; s < 8; ++s) { a[s] = lo[s]; a[s + 8] = hi[s]; }
                    afr[mt][kb] = a;
                }
            }
        }
        #pragma unroll
        for (int i = 0; i < TPW; ++i) {
            int nt = wave + i * NWAVE;
            if (nt < NT) {
                float bn = bias_r[i];
                v8f acc0 = {bn, bn, bn, bn, bn, bn, bn, bn};
                v8f acc1 = acc0;
                #pragma unroll
                for (int kb = 0; kb < KB; ++kb) {
                    acc0 = __builtin_amdgcn_wmma_f32_16x16x32_bf16(
                             false, afr[0][kb], false, breg[i][kb], (short)0, acc0, false, false);
                    acc1 = __builtin_amdgcn_wmma_f32_16x16x32_bf16(
                             false, afr[1][kb], false, breg[i][kb], (short)0, acc1, false, false);
                }
                *(v8f*)&pre[(nt * MT + 0) * 256 + lane * 8] = acc0;
                *(v8f*)&pre[(nt * MT + 1) * 256 + lane * 8] = acc1;
            }
        }
        __syncthreads();

        // ---- elementwise LSTM update + gather x(t+1)/mask(t+1) ----
        const int* mcur = &maskS[(t & 1) * TILE_B];
        #pragma unroll
        for (int it = 0; it < EPT; ++it) {
            int e = it * THREADS + tid;
            if (e < TILE_B * DIM_H) {
                int r = e / DIM_H, j = e % DIM_H;
                int mt = r >> 4, rr = r & 15;
                int lofs = (rr & 8) << 1;      // +16 lanes if rr >= 8
                int v    = rr & 7;
                int n0 = j, n1 = 100 + j, n2 = 200 + j, n3 = 300 + j;
                float pf = pre[((n0 >> 4) * MT + mt) * 256 + (((n0 & 15) + lofs) << 3) + v];
                float pi = pre[((n1 >> 4) * MT + mt) * 256 + (((n1 & 15) + lofs) << 3) + v];
                float po = pre[((n2 >> 4) * MT + mt) * 256 + (((n2 & 15) + lofs) << 3) + v];
                float pg = pre[((n3 >> 4) * MT + mt) * 256 + (((n3 & 15) + lofs) << 3) + v];
                float fg = sigmoid_hw(pf);
                float ig = sigmoid_hw(pi);
                float og = sigmoid_hw(po);
                float gg = tanh_hw(pg);
                float cn = c_reg[it] * fg + ig * gg;
                float hn = og * tanh_hw(cn);
                if (mcur[r]) {
                    c_reg[it] = cn;
                    h_reg[it] = hn;
                    Asta[r * KP + DIM_E + j] = (__bf16)hn;   // next step's A
                }
            }
        }
        if (t + 1 < T) {
            int* mnxt = &maskS[((t + 1) & 1) * TILE_B];
            if (tid < TILE_B) {
                int b = b0 + tid; if (b >= B) b = 0;
                int tok = data[(long)b * T + (t + 1)];
                mnxt[tid] = (tok != lastS[tid]);
                if (t + 2 < T) __builtin_prefetch(&data[(long)b * T + t + 2], 0, 3);
            }
            for (int idx = tid; idx < TILE_B * DIM_E; idx += THREADS) {
                int r = idx >> 5, c = idx & 31;
                int b = b0 + r; if (b >= B) b = 0;
                int tok = data[(long)b * T + (t + 1)];
                Asta[r * KP + c] = (__bf16)emb[tok * DIM_E + c];
            }
        }
        __syncthreads();
    }

    // ------------------------- output head -------------------------
    #pragma unroll
    for (int it = 0; it < EPT; ++it) {
        int e = it * THREADS + tid;
        if (e < TILE_B * DIM_H) {
            int r = e / DIM_H, j = e % DIM_H;
            atomicAdd(&outAcc[r], h_reg[it] * Wy[j]);
        }
    }
    __syncthreads();
    if (tid < TILE_B) {
        int b = b0 + tid;
        if (b < B) out[b] = sigmoid_hw(outAcc[tid] + by[0]);
    }
}

extern "C" void kernel_launch(void* const* d_in, const int* in_sizes, int n_in,
                              void* d_out, int out_size, void* d_ws, size_t ws_size,
                              hipStream_t stream) {
    const int*   data = (const int*)  d_in[0];
    const float* emb  = (const float*)d_in[1];
    const float* Wf   = (const float*)d_in[2];
    const float* bf   = (const float*)d_in[3];
    const float* Wi   = (const float*)d_in[4];
    const float* bi   = (const float*)d_in[5];
    const float* Wo   = (const float*)d_in[6];
    const float* bo   = (const float*)d_in[7];
    const float* Wc   = (const float*)d_in[8];
    const float* bc   = (const float*)d_in[9];
    const float* Wy   = (const float*)d_in[10];
    const float* by   = (const float*)d_in[11];
    float* out = (float*)d_out;

    int B = out_size;                 // output is (B, 1)
    int T = in_sizes[0] / B;          // data is (B, T)
    int nBlocks = (B + TILE_B - 1) / TILE_B;

    size_t smem = (size_t)NT * KB * 512 * sizeof(__bf16)   // Wpack
                + (size_t)TILE_B * KP * sizeof(__bf16)     // A staging
                + (size_t)NT * MT * 256 * sizeof(float)    // pre (fragment order)
                + (size_t)NGATE4 * sizeof(float)           // biases
                + (size_t)TILE_B * sizeof(float)           // out accum
                + (size_t)TILE_B * sizeof(int) * 3;        // mask x2 + last tokens

    hipLaunchKernelGGL(lstm_fused_wmma, dim3(nBlocks), dim3(THREADS), smem, stream,
                       data, emb, Wf, bf, Wi, bi, Wo, bo, Wc, bc, Wy, by,
                       out, B, T);
}
